// deStationaryFromer_69672959476045
// MI455X (gfx1250) — matrix operand order repaired and checked by reading, not compiled, not run on hardware
//
#include <hip/hip_runtime.h>
#include <hip/hip_bf16.h>
#include <cstdint>

// ---------------- problem constants ----------------
constexpr int Bc  = 32;
constexpr int Sc  = 512;
constexpr int Dc  = 512;
constexpr int Hc  = 8;
constexpr int NLc = 6;
constexpr int DFc = 2048;
constexpr int Ec  = Dc / Hc;      // 64
constexpr int Mtok = Bc * Sc;     // 16384 tokens
constexpr float EPSc = 1e-5f;

typedef _Float16 v8h  __attribute__((ext_vector_type(8)));
typedef _Float16 v16h __attribute__((ext_vector_type(16)));
typedef float    v8f  __attribute__((ext_vector_type(8)));
typedef unsigned int u32x4 __attribute__((ext_vector_type(4)));
typedef int          i32x4 __attribute__((ext_vector_type(4)));
typedef int          i32x8 __attribute__((ext_vector_type(8)));

#if defined(__gfx1250__) && defined(__has_builtin)
#if __has_builtin(__builtin_amdgcn_tensor_load_to_lds) && \
    __has_builtin(__builtin_amdgcn_s_wait_tensorcnt)
#define USE_TDM 1
#endif
#endif
#ifndef USE_TDM
#define USE_TDM 0
#endif

__device__ __forceinline__ v16h cat8(v8h lo, v8h hi) {
  return __builtin_shufflevector(lo, hi, 0,1,2,3,4,5,6,7,8,9,10,11,12,13,14,15);
}

__device__ __forceinline__ float gelu_exact(float x) {
  return 0.5f * x * (1.0f + erff(x * 0.70710678118654752f));
}

// ---------------- f32 -> f16 convert ----------------
__global__ void f32_to_f16_kernel(const float* __restrict__ src,
                                  _Float16* __restrict__ dst, int n) {
  int i = blockIdx.x * blockDim.x + threadIdx.x;
  if (i < n) dst[i] = (_Float16)src[i];
}

// ---------------- generic WMMA GEMM: C = A(MxK) * W(NxK)^T + bias ---------
// 128 threads = 4 waves; block tile 64x64; wave tile 32x32 (2x2 WMMA frags).
// A staged through LDS in 64x64 K-chunks, double buffered; on gfx1250 the
// staging is done by the Tensor Data Mover (tensor_load_to_lds + TENSORcnt).
enum { EPI_F32 = 0, EPI_F16 = 1, EPI_GELU_F16 = 2, EPI_KVT = 3 };

template <int EPI>
__global__ __launch_bounds__(128)
void gemm_wmma_kernel(const _Float16* __restrict__ A,
                      const _Float16* __restrict__ W,
                      const float* __restrict__ bias,
                      void* __restrict__ outp,
                      int M, int N, int K) {
  __shared__ __align__(16) _Float16 As[2][64 * 64];   // 2 x 8KB
  const int tid   = threadIdx.x;
  const int lane  = tid & 31;
  const int wave  = tid >> 5;
  const int m0    = blockIdx.x * 64;
  const int n0    = blockIdx.y * 64;
  const int wr    = (wave >> 1) * 32;   // wave row quadrant
  const int wc    = (wave & 1) * 32;    // wave col quadrant

  const int mA    = lane & 15;
  const int nlane = lane & 15;
  const int khalf = lane >> 4;
  const int kloA  = khalf * 16;
  const int kloB  = khalf * 16;

  const _Float16* wrow0 = W + (size_t)(n0 + wc + nlane) * K;
  const _Float16* wrow1 = W + (size_t)(n0 + wc + 16 + nlane) * K;

  const int nchunks = K >> 6;

#if USE_TDM
  // Tensor Data Mover staging of one 64x64 f16 tile (rows m0..m0+63,
  // K-cols chunk*64..+63) into As[buf]. D# per cdna5 ISA 08_async_tensor §8.
  auto tdm_issue = [&](int chunk, int buf) {
    uint64_t gaddr = (uint64_t)(uintptr_t)(A + (size_t)m0 * K + chunk * 64);
    unsigned  lds  = (unsigned)(uintptr_t)&As[buf][0];
    u32x4 g0;
    g0[0] = 1u;                                    // count=1, user mode
    g0[1] = lds;                                   // lds_addr
    g0[2] = (unsigned)(gaddr & 0xffffffffu);       // global_addr[31:0]
    g0[3] = (unsigned)((gaddr >> 32) & 0x01ffffffu) | 0x80000000u; // type=2
    i32x8 g1;
    g1[0] = 1 << 16;                               // data_size = 2 bytes
    g1[1] = (int)((K & 0xffff) << 16);             // tensor_dim0[15:0]
    g1[2] = (K >> 16) | ((M & 0xffff) << 16);      // dim0 hi | tensor_dim1 lo
    g1[3] = (M >> 16) | (64 << 16);                // dim1 hi | tile_dim0=64
    g1[4] = 64;                                    // tile_dim1=64, tile_dim2=0
    g1[5] = K;                                     // tensor_dim0_stride[31:0]
    g1[6] = 0;
    g1[7] = 0;
    i32x4 z4 = {0, 0, 0, 0};
    i32x8 z8 = {0, 0, 0, 0, 0, 0, 0, 0};
    __builtin_amdgcn_tensor_load_to_lds(g0, g1, z4, z4, z8, 0);
  };
  if (wave == 0) {
    tdm_issue(0, 0);
    __builtin_amdgcn_s_wait_tensorcnt(0);
  }
  __syncthreads();
#else
  auto stage = [&](int chunk, int buf) {
    const _Float16* src = A + (size_t)m0 * K + chunk * 64;
    for (int idx = tid * 8; idx < 64 * 64; idx += 128 * 8) {
      int r = idx >> 6, cc = idx & 63;
      *(v8h*)&As[buf][idx] = *(const v8h*)&src[(size_t)r * K + cc];
    }
  };
  stage(0, 0);
  __syncthreads();
#endif

  v8f a00 = {}, a01 = {}, a10 = {}, a11 = {};
  for (int c = 0; c < nchunks; ++c) {
    const int buf = c & 1;
    const int kc  = c << 6;
#if USE_TDM
    if (wave == 0 && c + 1 < nchunks) tdm_issue(c + 1, buf ^ 1);
#else
    if (c + 1 < nchunks) stage(c + 1, buf ^ 1);
#endif
    __builtin_prefetch(wrow0 + kc + 256, 0, 1);    // global_prefetch_b8
    __builtin_prefetch(wrow1 + kc + 256, 0, 1);
#pragma unroll
    for (int kk = 0; kk < 64; kk += 32) {
      const _Float16* ar0 = &As[buf][(wr + mA) * 64 + kk + kloA];
      const _Float16* ar1 = &As[buf][(wr + 16 + mA) * 64 + kk + kloA];
      v16h fa0 = cat8(*(const v8h*)ar0, *(const v8h*)(ar0 + 8));
      v16h fa1 = cat8(*(const v8h*)ar1, *(const v8h*)(ar1 + 8));
      v16h fb0 = *(const v16h*)&wrow0[kc + kk + kloB];
      v16h fb1 = *(const v16h*)&wrow1[kc + kk + kloB];
      a00 = __builtin_amdgcn_wmma_f32_16x16x32_f16(false, fa0, false, fb0,
                                                   (short)0, a00, false, false);
      a01 = __builtin_amdgcn_wmma_f32_16x16x32_f16(false, fa0, false, fb1,
                                                   (short)0, a01, false, false);
      a10 = __builtin_amdgcn_wmma_f32_16x16x32_f16(false, fa1, false, fb0,
                                                   (short)0, a10, false, false);
      a11 = __builtin_amdgcn_wmma_f32_16x16x32_f16(false, fa1, false, fb1,
                                                   (short)0, a11, false, false);
    }
#if USE_TDM
    if (wave == 0) __builtin_amdgcn_s_wait_tensorcnt(0);
#endif
    __syncthreads();
  }

  auto store_tile = [&](v8f acc, int rt, int ct) {
    const int nB = n0 + ct + nlane;
    const float bv = bias ? bias[nB] : 0.0f;
    const int rbase = m0 + rt + (khalf << 3);
#pragma unroll
    for (int v = 0; v < 8; ++v) {
      int row = rbase + v;
      float val = acc[v] + bv;
      if (EPI == EPI_GELU_F16) val = gelu_exact(val);
      if (EPI == EPI_F32) {
        ((float*)outp)[(size_t)row * N + nB] = val;
      } else if (EPI == EPI_F16 || EPI == EPI_GELU_F16) {
        ((_Float16*)outp)[(size_t)row * N + nB] = (_Float16)val;
      } else {  // EPI_KVT: scatter to (B,H,E,S) so P*V B-frags are contiguous
        int b = row / Sc, s = row - b * Sc;
        int h = nB / Ec, e = nB - h * Ec;
        ((_Float16*)outp)[(((size_t)b * Hc + h) * Ec + e) * Sc + s] =
            (_Float16)val;
      }
    }
  };
  store_tile(a00, wr, wc);
  store_tile(a01, wr, wc + 16);
  store_tile(a10, wr + 16, wc);
  store_tile(a11, wr + 16, wc + 16);
}

// ---------------- fused attention (flash-style online softmax) -----------
// grid: B*H*(S/64); 4 independent waves per block, 16 query rows per wave.
// Row-sum of the probability tile is computed with a WMMA against an
// all-ones B matrix (C layout replicates the row sum across the lane half).
__global__ __launch_bounds__(128)
void attn_wmma_kernel(const _Float16* __restrict__ qh,   // (B,S,D)
                      const _Float16* __restrict__ kh,   // (B,S,D)
                      const _Float16* __restrict__ vth,  // (B,H,E,S)
                      const float* __restrict__ tau,     // (B)
                      const float* __restrict__ delta,   // (B,S)
                      _Float16* __restrict__ attnh) {    // (B,S,D)
  __shared__ __align__(16) _Float16 pbuf[4][16 * 32];    // per-wave scratch
  const int tid = threadIdx.x, lane = tid & 31, wave = tid >> 5;
  int blk = blockIdx.x;
  const int qt = blk & (Sc / 64 - 1); blk >>= 3;   // S/64 == 8
  const int h  = blk & (Hc - 1);
  const int b  = blk >> 3;
  const int m0 = qt * 64 + wave * 16;

  const float tbs = tau[b] * 0.125f;               // tau * 1/sqrt(E)

  const _Float16* qbase = qh + (size_t)b * Sc * Dc + h * Ec;
  const _Float16* kbase = kh + (size_t)b * Sc * Dc + h * Ec;
  const _Float16* vbase = vth + ((size_t)b * Hc + h) * Ec * Sc;
  const float*    dbase = delta + (size_t)b * Sc;

  const int mA = lane & 15, nlane = lane & 15, khalf = lane >> 4;
  const int kloA = khalf * 16;

  v16h vones;
#pragma unroll
  for (int i = 0; i < 16; ++i) vones[i] = (_Float16)1.0f;

  // query A-fragments for E-steps kk=0,32
  v16h qa0, qa1;
  {
    const _Float16* qr = qbase + (size_t)(m0 + mA) * Dc;
    qa0 = cat8(*(const v8h*)&qr[kloA],      *(const v8h*)&qr[8 + kloA]);
    qa1 = cat8(*(const v8h*)&qr[32 + kloA], *(const v8h*)&qr[40 + kloA]);
  }

  v8f o0 = {}, o1 = {}, o2 = {}, o3 = {};
  float rm[8], rl[8];
#pragma unroll
  for (int v = 0; v < 8; ++v) { rm[v] = -1e30f; rl[v] = 0.0f; }

  for (int jt = 0; jt < Sc; jt += 32) {
    // scores for two 16-key sub-tiles
    v8f sc0 = {}, sc1 = {};
    {
      const _Float16* kr0 = kbase + (size_t)(jt + nlane) * Dc;
      v16h kb0 = *(const v16h*)&kr0[khalf * 16];
      v16h kb1 = *(const v16h*)&kr0[32 + khalf * 16];
      sc0 = __builtin_amdgcn_wmma_f32_16x16x32_f16(false, qa0, false, kb0,
                                                   (short)0, sc0, false, false);
      sc0 = __builtin_amdgcn_wmma_f32_16x16x32_f16(false, qa1, false, kb1,
                                                   (short)0, sc0, false, false);
      const _Float16* kr1 = kbase + (size_t)(jt + 16 + nlane) * Dc;
      v16h kc0 = *(const v16h*)&kr1[khalf * 16];
      v16h kc1 = *(const v16h*)&kr1[32 + khalf * 16];
      sc1 = __builtin_amdgcn_wmma_f32_16x16x32_f16(false, qa0, false, kc0,
                                                   (short)0, sc1, false, false);
      sc1 = __builtin_amdgcn_wmma_f32_16x16x32_f16(false, qa1, false, kc1,
                                                   (short)0, sc1, false, false);
    }
    const float d0 = dbase[jt + nlane] * 0.125f;
    const float d1 = dbase[jt + 16 + nlane] * 0.125f;
#pragma unroll
    for (int v = 0; v < 8; ++v) {
      float sa = sc0[v] * tbs + d0;
      float sb = sc1[v] * tbs + d1;
      // row max over 32 keys: pairwise max then 16-lane-half butterfly
      float t = fmaxf(sa, sb);
      t = fmaxf(t, __shfl_xor(t, 1, 32));
      t = fmaxf(t, __shfl_xor(t, 2, 32));
      t = fmaxf(t, __shfl_xor(t, 4, 32));
      t = fmaxf(t, __shfl_xor(t, 8, 32));
      float mn = fmaxf(rm[v], t);
      float f  = __expf(rm[v] - mn);
      rm[v] = mn;
      int prow = (v + (khalf << 3)) * 32;
      pbuf[wave][prow + nlane]      = (_Float16)__expf(sa - mn);
      pbuf[wave][prow + 16 + nlane] = (_Float16)__expf(sb - mn);
      o0[v] *= f; o1[v] *= f; o2[v] *= f; o3[v] *= f;
      rl[v] *= f;
    }
    // per-wave LDS ops are architecturally in-order: no barrier needed
    v16h pa;
    {
      const _Float16* pr = &pbuf[wave][mA * 32];
      pa = cat8(*(const v8h*)&pr[kloA], *(const v8h*)&pr[8 + kloA]);
    }
    // row-sum via WMMA against ones
    v8f rs = {};
    rs = __builtin_amdgcn_wmma_f32_16x16x32_f16(false, pa, false, vones,
                                                (short)0, rs, false, false);
#pragma unroll
    for (int v = 0; v < 8; ++v) rl[v] += rs[v];
    // P @ V
#pragma unroll
    for (int et = 0; et < 4; ++et) {
      const _Float16* vr =
          vbase + (size_t)(et * 16 + nlane) * Sc + jt + khalf * 16;
      v16h vb = *(const v16h*)vr;
      v8f* op = (et == 0) ? &o0 : (et == 1) ? &o1 : (et == 2) ? &o2 : &o3;
      *op = __builtin_amdgcn_wmma_f32_16x16x32_f16(false, pa, false, vb,
                                                   (short)0, *op, false, false);
    }
  }

  _Float16* ob = attnh + (size_t)b * Sc * Dc + h * Ec;
#pragma unroll
  for (int v = 0; v < 8; ++v) {
    int row = m0 + v + (khalf << 3);
    float inv = 1.0f / rl[v];
    ob[(size_t)row * Dc +  0 + nlane] = (_Float16)(o0[v] * inv);
    ob[(size_t)row * Dc + 16 + nlane] = (_Float16)(o1[v] * inv);
    ob[(size_t)row * Dc + 32 + nlane] = (_Float16)(o2[v] * inv);
    ob[(size_t)row * Dc + 48 + nlane] = (_Float16)(o3[v] * inv);
  }
}

// ---------------- LayerNorm (optionally fused residual add) --------------
__global__ __launch_bounds__(128)
void ln_kernel(const float* __restrict__ xin, const float* __restrict__ addin,
               const float* __restrict__ gamma, const float* __restrict__ beta,
               float* __restrict__ xoutf, _Float16* __restrict__ xouth) {
  __shared__ float sa[128], sb[128];
  const int t = blockIdx.x;
  const float* xr = xin + (size_t)t * Dc;
  const float* ar = addin ? addin + (size_t)t * Dc : nullptr;
  float vals[4], s = 0.f, s2 = 0.f;
#pragma unroll
  for (int i = 0; i < 4; ++i) {
    int d = threadIdx.x * 4 + i;
    float x = xr[d] + (ar ? ar[d] : 0.f);
    vals[i] = x; s += x; s2 += x * x;
  }
  sa[threadIdx.x] = s; sb[threadIdx.x] = s2;
  __syncthreads();
  for (int off = 64; off > 0; off >>= 1) {
    if (threadIdx.x < off) {
      sa[threadIdx.x] += sa[threadIdx.x + off];
      sb[threadIdx.x] += sb[threadIdx.x + off];
    }
    __syncthreads();
  }
  float mean = sa[0] / Dc;
  float var  = sb[0] / Dc - mean * mean;
  float inv  = rsqrtf(var + EPSc);
#pragma unroll
  for (int i = 0; i < 4; ++i) {
    int d = threadIdx.x * 4 + i;
    float y = (vals[i] - mean) * inv * gamma[d] + beta[d];
    xoutf[(size_t)t * Dc + d] = y;
    xouth[(size_t)t * Dc + d] = (_Float16)y;
  }
}

// ---------------- final: LN -> GELU -> dot(Wout) + bout ------------------
__global__ __launch_bounds__(128)
void final_kernel(const float* __restrict__ xin, const float* __restrict__ gn,
                  const float* __restrict__ bn, const float* __restrict__ Wout,
                  const float* __restrict__ bout, float* __restrict__ out) {
  __shared__ float sa[128], sb[128];
  const int t = blockIdx.x;
  const float* xr = xin + (size_t)t * Dc;
  float vals[4], s = 0.f, s2 = 0.f;
#pragma unroll
  for (int i = 0; i < 4; ++i) {
    int d = threadIdx.x * 4 + i;
    float x = xr[d];
    vals[i] = x; s += x; s2 += x * x;
  }
  sa[threadIdx.x] = s; sb[threadIdx.x] = s2;
  __syncthreads();
  for (int off = 64; off > 0; off >>= 1) {
    if (threadIdx.x < off) {
      sa[threadIdx.x] += sa[threadIdx.x + off];
      sb[threadIdx.x] += sb[threadIdx.x + off];
    }
    __syncthreads();
  }
  float mean = sa[0] / Dc;
  float var  = sb[0] / Dc - mean * mean;
  float inv  = rsqrtf(var + EPSc);
  __syncthreads();
  float dot = 0.f;
#pragma unroll
  for (int i = 0; i < 4; ++i) {
    int d = threadIdx.x * 4 + i;
    float y = (vals[i] - mean) * inv * gn[d] + bn[d];
    dot += gelu_exact(y) * Wout[d];
  }
  sa[threadIdx.x] = dot;
  __syncthreads();
  for (int off = 64; off > 0; off >>= 1) {
    if (threadIdx.x < off) sa[threadIdx.x] += sa[threadIdx.x + off];
    __syncthreads();
  }
  if (threadIdx.x == 0) out[t] = sa[0] + bout[0];
}

// ---------------- embedding: circular conv + positional encoding ---------
__global__ void embed_kernel(const float* __restrict__ xe,
                             const float* __restrict__ Wemb,
                             float* __restrict__ xf, _Float16* __restrict__ xh) {
  int idx = blockIdx.x * blockDim.x + threadIdx.x;
  if (idx >= Bc * Sc * Dc) return;
  int d = idx % Dc;
  int s = (idx / Dc) % Sc;
  int b = idx / (Dc * Sc);
  int sm = (s == 0) ? Sc - 1 : s - 1;
  int sp = (s == Sc - 1) ? 0 : s + 1;
  const float* xb = xe + (size_t)b * Sc * 2;
  float acc = 0.f;
#pragma unroll
  for (int c = 0; c < 2; ++c) {
    const float* w = Wemb + (size_t)d * 6 + c * 3;
    acc += xb[sm * 2 + c] * w[0] + xb[s * 2 + c] * w[1] + xb[sp * 2 + c] * w[2];
  }
  int i2 = (d >> 1) << 1;
  float div = __expf(-logf(10000.0f) * (float)i2 / (float)Dc);
  float ang = (float)s * div;
  float pe = (d & 1) ? cosf(ang) : sinf(ang);
  float v = acc + pe;
  xf[idx] = v;
  xh[idx] = (_Float16)v;
}

// ---------------- projector stage 1: channel-conv + per-batch stats ------
__global__ __launch_bounds__(128)
void proj_pre_kernel(const float* __restrict__ xe, const float* __restrict__ Wtc,
                     const float* __restrict__ Wdc, float* __restrict__ featT,
                     float* __restrict__ featD) {
  __shared__ float red[8][128];
  const int b = blockIdx.x, tid = threadIdx.x;
  const float* xb = xe + (size_t)b * Sc * 2;
  float s0 = 0, s1 = 0, q0 = 0, q1 = 0, t0 = 0, t1 = 0, d0 = 0, d1 = 0;
  for (int s = tid; s < Sc; s += 128) {
    float x0 = xb[s * 2 + 0], x1 = xb[s * 2 + 1];
    s0 += x0; s1 += x1; q0 += x0 * x0; q1 += x1 * x1;
    float wt0 = Wtc[s * 3 + 0], wt1 = Wtc[s * 3 + 1], wt2 = Wtc[s * 3 + 2];
    t0 += x1 * wt0 + x0 * wt1 + x1 * wt2;   // circular pad [x1,x0,x1,x0]
    t1 += x0 * wt0 + x1 * wt1 + x0 * wt2;
    float wd0 = Wdc[s * 3 + 0], wd1 = Wdc[s * 3 + 1], wd2 = Wdc[s * 3 + 2];
    d0 += x1 * wd0 + x0 * wd1 + x1 * wd2;
    d1 += x0 * wd0 + x1 * wd1 + x0 * wd2;
  }
  red[0][tid] = s0; red[1][tid] = s1; red[2][tid] = q0; red[3][tid] = q1;
  red[4][tid] = t0; red[5][tid] = t1; red[6][tid] = d0; red[7][tid] = d1;
  __syncthreads();
  for (int off = 64; off > 0; off >>= 1) {
    if (tid < off)
      for (int j = 0; j < 8; ++j) red[j][tid] += red[j][tid + off];
    __syncthreads();
  }
  if (tid == 0) {
    float m0 = red[0][0] / Sc, m1 = red[1][0] / Sc;
    float v0 = red[2][0] / Sc - m0 * m0, v1 = red[3][0] / Sc - m1 * m1;
    featT[b * 4 + 0] = red[4][0]; featT[b * 4 + 1] = red[5][0];
    featT[b * 4 + 2] = sqrtf(v0 + EPSc); featT[b * 4 + 3] = sqrtf(v1 + EPSc);
    featD[b * 4 + 0] = red[6][0]; featD[b * 4 + 1] = red[7][0];
    featD[b * 4 + 2] = m0; featD[b * 4 + 3] = m1;
  }
}

// ---------------- projector stage 2: MLPs -> tau (B), delta (B,S) --------
__global__ __launch_bounds__(128)
void proj_mlp_kernel(const float* __restrict__ featT, const float* __restrict__ featD,
                     const float* __restrict__ Wt1, const float* __restrict__ bt1,
                     const float* __restrict__ Wt2, const float* __restrict__ bt2,
                     const float* __restrict__ Wt3,
                     const float* __restrict__ Wd1, const float* __restrict__ bd1,
                     const float* __restrict__ Wd2, const float* __restrict__ bd2,
                     const float* __restrict__ Wd3,
                     float* __restrict__ tau, float* __restrict__ delta) {
  __shared__ float h1t[128], h1d[128], h2t[128], h2d[128], red[128];
  const int b = blockIdx.x, i = threadIdx.x;
  const float* ft = featT + b * 4;
  const float* fd = featD + b * 4;
  float a = bt1[i], c = bd1[i];
#pragma unroll
  for (int j = 0; j < 4; ++j) {
    a += ft[j] * Wt1[i * 4 + j];
    c += fd[j] * Wd1[i * 4 + j];
  }
  h1t[i] = fmaxf(a, 0.f); h1d[i] = fmaxf(c, 0.f);
  __syncthreads();
  float a2 = bt2[i], c2 = bd2[i];
  for (int j = 0; j < 128; ++j) {
    a2 += h1t[j] * Wt2[i * 128 + j];
    c2 += h1d[j] * Wd2[i * 128 + j];
  }
  h2t[i] = fmaxf(a2, 0.f); h2d[i] = fmaxf(c2, 0.f);
  __syncthreads();
  red[i] = h2t[i] * Wt3[i];
  __syncthreads();
  for (int off = 64; off > 0; off >>= 1) {
    if (i < off) red[i] += red[i + off];
    __syncthreads();
  }
  if (i == 0) tau[b] = __expf(red[0]);
  for (int s = i; s < Sc; s += 128) {
    float acc = 0.f;
    for (int j = 0; j < 128; ++j) acc += h2d[j] * Wd3[s * 128 + j];
    delta[b * Sc + s] = acc;
  }
}

// ============================ host launcher ==============================
extern "C" void kernel_launch(void* const* d_in, const int* in_sizes, int n_in,
                              void* d_out, int out_size, void* d_ws, size_t ws_size,
                              hipStream_t stream) {
  (void)in_sizes; (void)n_in; (void)out_size; (void)ws_size;
  const float* x_enc = (const float*)d_in[0];
  const float* W_emb = (const float*)d_in[1];
  const float* Wq = (const float*)d_in[2];  const float* bq = (const float*)d_in[3];
  const float* Wk = (const float*)d_in[4];  const float* bk = (const float*)d_in[5];
  const float* Wv = (const float*)d_in[6];  const float* bv = (const float*)d_in[7];
  const float* Wo = (const float*)d_in[8];  const float* bo = (const float*)d_in[9];
  const float* Wc1 = (const float*)d_in[10]; const float* bc1 = (const float*)d_in[11];
  const float* Wc2 = (const float*)d_in[12]; const float* bc2 = (const float*)d_in[13];
  const float* g1w = (const float*)d_in[14]; const float* be1 = (const float*)d_in[15];
  const float* g2w = (const float*)d_in[16]; const float* be2 = (const float*)d_in[17];
  const float* gn = (const float*)d_in[18];  const float* bnv = (const float*)d_in[19];
  const float* Wout = (const float*)d_in[20]; const float* bout = (const float*)d_in[21];
  const float* Wtc = (const float*)d_in[22];
  const float* Wt1 = (const float*)d_in[23]; const float* bt1 = (const float*)d_in[24];
  const float* Wt2 = (const float*)d_in[25]; const float* bt2 = (const float*)d_in[26];
  const float* Wt3 = (const float*)d_in[27];
  const float* Wdc = (const float*)d_in[28];
  const float* Wd1 = (const float*)d_in[29]; const float* bd1 = (const float*)d_in[30];
  const float* Wd2 = (const float*)d_in[31]; const float* bd2 = (const float*)d_in[32];
  const float* Wd3 = (const float*)d_in[33];

  char* ws = (char*)d_ws;
  auto alloc = [&](size_t bytes) -> char* {
    char* p = ws;
    ws += (bytes + 255) & ~(size_t)255;
    return p;
  };
  float*    xf32  = (float*)alloc((size_t)Mtok * Dc * 4);
  _Float16* xh    = (_Float16*)alloc((size_t)Mtok * Dc * 2);
  _Float16* qh    = (_Float16*)alloc((size_t)Mtok * Dc * 2);
  _Float16* kh    = (_Float16*)alloc((size_t)Mtok * Dc * 2);
  _Float16* vth   = (_Float16*)alloc((size_t)Mtok * Dc * 2);
  _Float16* attnh = (_Float16*)alloc((size_t)Mtok * Dc * 2);
  float*    tmpf  = (float*)alloc((size_t)Mtok * Dc * 4);
  _Float16* ffnh  = (_Float16*)alloc((size_t)Mtok * DFc * 2);
  _Float16* whq   = (_Float16*)alloc((size_t)NLc * Dc * Dc * 2);
  _Float16* whk   = (_Float16*)alloc((size_t)NLc * Dc * Dc * 2);
  _Float16* whv   = (_Float16*)alloc((size_t)NLc * Dc * Dc * 2);
  _Float16* who   = (_Float16*)alloc((size_t)NLc * Dc * Dc * 2);
  _Float16* whc1  = (_Float16*)alloc((size_t)NLc * DFc * Dc * 2);
  _Float16* whc2  = (_Float16*)alloc((size_t)NLc * Dc * DFc * 2);
  float*    featT = (float*)alloc(Bc * 4 * 4);
  float*    featD = (float*)alloc(Bc * 4 * 4);
  float*    tau   = (float*)alloc(Bc * 4);
  float*    delta = (float*)alloc((size_t)Bc * Sc * 4);

  const int nqkv = NLc * Dc * Dc;
  const int nffn = NLc * DFc * Dc;
  f32_to_f16_kernel<<<(nqkv + 255) / 256, 256, 0, stream>>>(Wq, whq, nqkv);
  f32_to_f16_kernel<<<(nqkv + 255) / 256, 256, 0, stream>>>(Wk, whk, nqkv);
  f32_to_f16_kernel<<<(nqkv + 255) / 256, 256, 0, stream>>>(Wv, whv, nqkv);
  f32_to_f16_kernel<<<(nqkv + 255) / 256, 256, 0, stream>>>(Wo, who, nqkv);
  f32_to_f16_kernel<<<(nffn + 255) / 256, 256, 0, stream>>>(Wc1, whc1, nffn);
  f32_to_f16_kernel<<<(nffn + 255) / 256, 256, 0, stream>>>(Wc2, whc2, nffn);

  proj_pre_kernel<<<Bc, 128, 0, stream>>>(x_enc, Wtc, Wdc, featT, featD);
  proj_mlp_kernel<<<Bc, 128, 0, stream>>>(featT, featD, Wt1, bt1, Wt2, bt2, Wt3,
                                          Wd1, bd1, Wd2, bd2, Wd3, tau, delta);
  embed_kernel<<<(Bc * Sc * Dc + 255) / 256, 256, 0, stream>>>(x_enc, W_emb,
                                                               xf32, xh);

  const dim3 gD(Mtok / 64, Dc / 64);    // (256, 8)
  const dim3 gF(Mtok / 64, DFc / 64);   // (256, 32)

  for (int l = 0; l < NLc; ++l) {
    const _Float16* wq_l = whq + (size_t)l * Dc * Dc;
    const _Float16* wk_l = whk + (size_t)l * Dc * Dc;
    const _Float16* wv_l = whv + (size_t)l * Dc * Dc;
    const _Float16* wo_l = who + (size_t)l * Dc * Dc;
    const _Float16* w1_l = whc1 + (size_t)l * DFc * Dc;
    const _Float16* w2_l = whc2 + (size_t)l * Dc * DFc;

    gemm_wmma_kernel<EPI_F16><<<gD, 128, 0, stream>>>(
        xh, wq_l, bq + l * Dc, qh, Mtok, Dc, Dc);
    gemm_wmma_kernel<EPI_F16><<<gD, 128, 0, stream>>>(
        xh, wk_l, bk + l * Dc, kh, Mtok, Dc, Dc);
    gemm_wmma_kernel<EPI_KVT><<<gD, 128, 0, stream>>>(
        xh, wv_l, bv + l * Dc, vth, Mtok, Dc, Dc);

    attn_wmma_kernel<<<Bc * Hc * (Sc / 64), 128, 0, stream>>>(
        qh, kh, vth, tau, delta, attnh);

    gemm_wmma_kernel<EPI_F32><<<gD, 128, 0, stream>>>(
        attnh, wo_l, bo + l * Dc, tmpf, Mtok, Dc, Dc);
    ln_kernel<<<Mtok, 128, 0, stream>>>(xf32, tmpf, g1w + l * Dc, be1 + l * Dc,
                                        xf32, xh);

    gemm_wmma_kernel<EPI_GELU_F16><<<gF, 128, 0, stream>>>(
        xh, w1_l, bc1 + l * DFc, ffnh, Mtok, DFc, Dc);
    gemm_wmma_kernel<EPI_F32><<<gD, 128, 0, stream>>>(
        ffnh, w2_l, bc2 + l * Dc, tmpf, Mtok, Dc, DFc);
    ln_kernel<<<Mtok, 128, 0, stream>>>(xf32, tmpf, g2w + l * Dc, be2 + l * Dc,
                                        xf32, xh);
  }

  final_kernel<<<Mtok, 128, 0, stream>>>(xf32, gn, bnv, Wout, bout,
                                         (float*)d_out);
}
